// QuantizedLinearLayer_6476810682794
// MI455X (gfx1250) — matrix operand order repaired and checked by reading, not compiled, not run on hardware
//
#include <hip/hip_runtime.h>
#include <stdint.h>

// ---- CDNA5 vector types -----------------------------------------------------
typedef __attribute__((ext_vector_type(16))) _Float16 v16h;
typedef __attribute__((ext_vector_type(8)))  _Float16 v8h;
typedef __attribute__((ext_vector_type(4)))  _Float16 v4h;
typedef __attribute__((ext_vector_type(2)))  _Float16 h2;
typedef __attribute__((ext_vector_type(8)))  float    v8f;

union H16 { v16h v; v8h h[2]; };
union V8U { v8h v; uint32_t u[4]; };

// ---- Problem constants (B=2, S=2048, DIN=4096, DOUT=4096) -------------------
#define M_DIM 4096   // B*S
#define N_DIM 4096   // DOUT
#define K_DIM 4096   // DIN

#define BM 128
#define BN 256
#define BK 32
#define LDT (BK + 8) // LDS row stride in halves: 40 halves = 80B (16B aligned)

// int8 -> f16 via magic-number trick:
//   u8 = i8 ^ 0x80 (bias to [0,255]); f16 bits (0x6400 | u8) == 1024.0h + u8
//   then subtract 1152 (=1024+128) with one packed op per half2.
__device__ __forceinline__ uint32_t sub1152_pk(uint32_t p) {
  union { uint32_t u; h2 h; } t;
  t.u = p;
  const h2 c = {(_Float16)1152.0f, (_Float16)1152.0f};
  t.h = t.h - c;             // v_pk_add_f16 (neg)
  return t.u;
}
__device__ __forceinline__ void cvt_i8x4_f16x4(uint32_t d, uint32_t& o0, uint32_t& o1) {
  const uint32_t u = d ^ 0x80808080u;                         // v_xor_b32
  // v_perm_b32: codes 0-3 pick S1 bytes (0x64), 4-7 pick S0 bytes (u)
  uint32_t a = __builtin_amdgcn_perm(u, 0x64646464u, 0x00050004u); // [64,u1,64,u0]
  uint32_t b = __builtin_amdgcn_perm(u, 0x64646464u, 0x00070006u); // [64,u3,64,u2]
  o0 = sub1152_pk(a);
  o1 = sub1152_pk(b);
}

__launch_bounds__(256)
__global__ void qlinear_wmma_f16_kernel(const float* __restrict__ x,
                                        const int8_t* __restrict__ w,
                                        const float* __restrict__ scale,
                                        const float* __restrict__ bias,
                                        float* __restrict__ out) {
  // Double-buffered staging tiles: 20KB (A) + 40KB (B) = 60KB < 320KB/WGP
  __shared__ _Float16 sA[2][BM][LDT];
  __shared__ _Float16 sB[2][BN][LDT];

  const int tid    = threadIdx.x;
  const int lane   = tid & 31;
  const int wid    = tid >> 5;     // 8 waves
  const int lane16 = lane & 15;
  const int laneHi = lane >> 4;    // 0 or 1

  const int wave_m = wid & 1;      // 2 waves along M -> 64 rows each
  const int wave_n = wid >> 1;     // 4 waves along N -> 64 cols each

  const int blockM = blockIdx.y * BM;
  const int blockN = blockIdx.x * BN;

  // ---- global->register staging state ---------------------------------------
  float4 aReg[4];    // 16 fp32 of x per thread per K-step (128x32 tile)
  int4   bReg[2];    // 32 int8 of weight per thread per K-step (256x32 tile)

  auto load_global = [&](int kt) {
#pragma unroll
    for (int i = 0; i < 4; ++i) {
      const int lin = tid + i * 256;          // 0..1023
      const int ar  = lin >> 3;               // 0..127
      const int ac4 = lin & 7;                // float4 index within 32-wide row
      aReg[i] = *(const float4*)&x[(blockM + ar) * K_DIM + kt * BK + ac4 * 4];
    }
#pragma unroll
    for (int i = 0; i < 2; ++i) {
      const int lin = tid + i * 256;          // 0..511
      const int br  = lin >> 1;               // 0..255
      const int bc  = (lin & 1) * 16;         // 0 or 16
      bReg[i] = *(const int4*)&w[(blockN + br) * K_DIM + kt * BK + bc];
    }
  };

  auto store_lds = [&](int buf) {
#pragma unroll
    for (int i = 0; i < 4; ++i) {
      const int lin = tid + i * 256;
      const int ar  = lin >> 3;
      const int ac4 = lin & 7;
      v4h h;
      h[0] = (_Float16)aReg[i].x;             // v_cvt_pk_f16_f32 pairs
      h[1] = (_Float16)aReg[i].y;
      h[2] = (_Float16)aReg[i].z;
      h[3] = (_Float16)aReg[i].w;
      *(v4h*)&sA[buf][ar][ac4 * 4] = h;       // 8B store, 8B aligned
    }
#pragma unroll
    for (int i = 0; i < 2; ++i) {
      const int lin = tid + i * 256;
      const int br  = lin >> 1;
      const int bc  = (lin & 1) * 16;
      V8U lo, hi;
      cvt_i8x4_f16x4((uint32_t)bReg[i].x, lo.u[0], lo.u[1]);
      cvt_i8x4_f16x4((uint32_t)bReg[i].y, lo.u[2], lo.u[3]);
      cvt_i8x4_f16x4((uint32_t)bReg[i].z, hi.u[0], hi.u[1]);
      cvt_i8x4_f16x4((uint32_t)bReg[i].w, hi.u[2], hi.u[3]);
      *(v8h*)&sB[buf][br][bc]     = lo.v;     // 16B store, 16B aligned
      *(v8h*)&sB[buf][br][bc + 8] = hi.v;
    }
  };

  // ---- accumulators: 4(M) x 4(N) tiles of 16x16 f32 = 128 VGPRs --------------
  v8f acc[4][4];
  const v8f vzero = {0.f, 0.f, 0.f, 0.f, 0.f, 0.f, 0.f, 0.f};
#pragma unroll
  for (int mt = 0; mt < 4; ++mt)
#pragma unroll
    for (int nt = 0; nt < 4; ++nt)
      acc[mt][nt] = vzero;

  const int nK = K_DIM / BK;   // 128 K-steps

  load_global(0);
  store_lds(0);
  __syncthreads();

  for (int kt = 0; kt < nK; ++kt) {
    const int buf = kt & 1;

    // Prefetch next tile into registers; hide HBM/L2 latency under WMMAs.
    if (kt + 1 < nK) {
      load_global(kt + 1);
      if (kt + 2 < nK)
        __builtin_prefetch(&w[(blockN + (tid >> 1)) * K_DIM + (kt + 2) * BK], 0, 3);
    }

    // ---- fragment loads from LDS (ISA 7.12.2 layouts) -----------------------
    const int kbA = laneHi * 8;   // A: lane holds K runs [kbA,kbA+8) and +16
    const int kbB = laneHi * 8;   // B: lane holds K runs [kbB,kbB+8) and +16

    H16 af[4], bf[4];
#pragma unroll
    for (int mt = 0; mt < 4; ++mt) {
      const _Float16* p = &sA[buf][wave_m * 64 + mt * 16 + lane16][0];
      af[mt].h[0] = *(const v8h*)(p + kbA);
      af[mt].h[1] = *(const v8h*)(p + kbA + 16);
    }
#pragma unroll
    for (int nt = 0; nt < 4; ++nt) {
      const _Float16* p = &sB[buf][wave_n * 64 + nt * 16 + lane16][0];
      bf[nt].h[0] = *(const v8h*)(p + kbB);
      bf[nt].h[1] = *(const v8h*)(p + kbB + 16);
    }

    // ---- 16x V_WMMA_F32_16X16X32_F16 ----------------------------------------
#pragma unroll
    for (int mt = 0; mt < 4; ++mt)
#pragma unroll
      for (int nt = 0; nt < 4; ++nt)
        acc[mt][nt] = __builtin_amdgcn_wmma_f32_16x16x32_f16(
            false, af[mt].v, false, bf[nt].v,
            (short)0, acc[mt][nt], false, false);

    // Fill the other buffer for the next iteration; one barrier per K-step.
    if (kt + 1 < nK) {
      store_lds(buf ^ 1);
      __syncthreads();
    }
  }

  // ---- epilogue: dequant scale + bias, C/D layout per ISA 7.12.2 ------------
#pragma unroll
  for (int nt = 0; nt < 4; ++nt) {
    const int   n  = blockN + wave_n * 64 + nt * 16 + lane16;
    const float sc = scale[n];
    const float bi = bias[n];
#pragma unroll
    for (int mt = 0; mt < 4; ++mt) {
      const int m0 = blockM + wave_m * 64 + mt * 16 + laneHi * 8;
#pragma unroll
      for (int r = 0; r < 8; ++r)
        out[(m0 + r) * N_DIM + n] = acc[mt][nt][r] * sc + bi;
    }
  }
}

extern "C" void kernel_launch(void* const* d_in, const int* in_sizes, int n_in,
                              void* d_out, int out_size, void* d_ws, size_t ws_size,
                              hipStream_t stream) {
  const float*  x     = (const float*)d_in[0];
  const int8_t* w     = (const int8_t*)d_in[1];   // int8 quantized weights
  const float*  scale = (const float*)d_in[2];
  const float*  bias  = (const float*)d_in[3];
  float*        out   = (float*)d_out;

  dim3 grid(N_DIM / BN, M_DIM / BM);   // 16 x 32 workgroups
  qlinear_wmma_f16_kernel<<<grid, 256, 0, stream>>>(x, w, scale, bias, out);

  (void)in_sizes; (void)n_in; (void)out_size; (void)d_ws; (void)ws_size;
}
